// StageARegistry_60112362274961
// MI455X (gfx1250) — matrix operand order repaired and checked by reading, not compiled, not run on hardware
//
#include <hip/hip_runtime.h>

// ---------------------------------------------------------------------------
// VQ nearest-codebook assignment for MI455X (gfx1250, wave32, WMMA).
// score = c_sq[k] - 2 * (z . c[k])  (||z||^2 is argmin-invariant).
// Cross term: split-precision bf16x3 GEMM on v_wmma_f32_16x16x32_bf16.
// Codebook chunks double-buffered in LDS via global_load_async_to_lds_b128
// (ASYNCcnt), overlapping DMA with WMMA compute.
// ---------------------------------------------------------------------------

typedef __attribute__((ext_vector_type(16))) __bf16 v16bf;
typedef __attribute__((ext_vector_type(8)))  __bf16 v8bf;
typedef __attribute__((ext_vector_type(8)))  float  v8f;

#define D_DIM     256
#define K_CODES   1024
#define M_TILE    128          // tokens per workgroup (8 waves x 16 tokens)
#define N_CHUNK   64           // codes staged in LDS per iteration
#define N_CHUNKS  (K_CODES / N_CHUNK)
#define CB_STRIDE 264          // bf16 elems per LDS codebook row (pad vs 256)

__device__ __forceinline__ v8f wmma_bf16(v16bf a, v16bf b, v8f c) {
  return __builtin_amdgcn_wmma_f32_16x16x32_bf16(false, a, false, b,
                                                 (short)0, c, false, false);
}

// Async DMA: global (16B/lane) -> LDS, tracked by ASYNCcnt.
__device__ __forceinline__ void async_load_b128(unsigned int lds_off,
                                                const void* gptr) {
  unsigned long long ga = (unsigned long long)(uintptr_t)gptr;
  asm volatile("global_load_async_to_lds_b128 %0, %1, off"
               :
               : "v"(lds_off), "v"(ga)
               : "memory");
}

__device__ __forceinline__ void wait_asynccnt0() {
  asm volatile("s_wait_asynccnt 0" ::: "memory");
}

// ---------------------------------------------------------------------------
// Kernel 1: codebook fp32 -> bf16 hi/lo split + c_sq per code.
// ---------------------------------------------------------------------------
__global__ __launch_bounds__(256) void vq_prep(
    const float* __restrict__ cb,     // [K, D] fp32
    __bf16* __restrict__ cb_hi,       // [K, D] bf16
    __bf16* __restrict__ cb_lo,       // [K, D] bf16
    float* __restrict__ csq_g) {      // [K]
  const int k = blockIdx.x;
  const int d = threadIdx.x;
  float x = cb[k * D_DIM + d];
  __bf16 h = (__bf16)x;
  float resid = x - (float)h;
  cb_hi[k * D_DIM + d] = h;
  cb_lo[k * D_DIM + d] = (__bf16)resid;

  float sq = x * x;
#pragma unroll
  for (int m = 16; m >= 1; m >>= 1) sq += __shfl_xor(sq, m, 32);
  __shared__ float ssum[8];
  if ((threadIdx.x & 31) == 0) ssum[threadIdx.x >> 5] = sq;
  __syncthreads();
  if (threadIdx.x == 0) {
    float t = 0.f;
#pragma unroll
    for (int i = 0; i < 8; ++i) t += ssum[i];
    csq_g[k] = t;
  }
}

// ---------------------------------------------------------------------------
// Kernel 2: main assignment + gather.  256 blocks x 256 threads (8 waves).
// ---------------------------------------------------------------------------
__global__ __launch_bounds__(256) void vq_assign(
    const float* __restrict__ z,       // [32768, 256] fp32
    const float* __restrict__ cb_f32,  // [1024, 256] fp32 (for gather)
    const __bf16* __restrict__ cb_hi,  // [1024, 256] bf16
    const __bf16* __restrict__ cb_lo,  // [1024, 256] bf16
    const float* __restrict__ csq_g,   // [1024]
    float* __restrict__ out) {         // [32768, 256] fp32
  __shared__ __align__(16) __bf16 s_hi[2][N_CHUNK * CB_STRIDE];
  __shared__ __align__(16) __bf16 s_lo[2][N_CHUNK * CB_STRIDE];
  __shared__ __align__(16) float s_csq[K_CODES];
  __shared__ int s_idx[M_TILE];

  const int tid   = threadIdx.x;
  const int wave  = tid >> 5;
  const int lane  = tid & 31;
  const int half  = lane >> 4;   // 0: lanes 0-15, 1: lanes 16-31
  const int mlane = lane & 15;
  const int tok_base = blockIdx.x * M_TILE;
  const int my_token = tok_base + wave * 16 + mlane;

  // Per-thread slice of each staged chunk: row r, 64 dims starting at c0.
  const int srow = tid >> 2;           // 0..63
  const int scol = (tid & 3) * 64;     // 0,64,128,192
  const unsigned int lds_hi0 =
      (unsigned int)(uintptr_t)&s_hi[0][srow * CB_STRIDE + scol];
  const unsigned int lds_hi1 =
      (unsigned int)(uintptr_t)&s_hi[1][srow * CB_STRIDE + scol];
  const unsigned int lds_lo0 =
      (unsigned int)(uintptr_t)&s_lo[0][srow * CB_STRIDE + scol];
  const unsigned int lds_lo1 =
      (unsigned int)(uintptr_t)&s_lo[1][srow * CB_STRIDE + scol];

  // ---- Issue async DMA for chunk 0 into buffer 0 (hides behind A load).
  {
    const __bf16* ghi = cb_hi + (size_t)srow * D_DIM + scol;
    const __bf16* glo = cb_lo + (size_t)srow * D_DIM + scol;
#pragma unroll
    for (int j = 0; j < 64; j += 8) {
      async_load_b128(lds_hi0 + j * 2, ghi + j);
      async_load_b128(lds_lo0 + j * 2, glo + j);
    }
  }
  // ---- Stage all 1024 c_sq values once (4 KB).
  *(float4*)(s_csq + tid * 4) = *(const float4*)(csq_g + tid * 4);

  // ---- Load this wave's 16x256 A tile into registers, split hi/lo bf16.
  // ISA 16-bit A 16x32 layout: lane holds row m=lane%16;
  // element e -> k = half*8 + (e<8 ? e : e+8) within each 32-wide K step.
  v16bf a_hi[8], a_lo[8];
  {
    const float* zrow = z + (size_t)my_token * D_DIM;
#pragma unroll
    for (int s = 0; s < 8; ++s) {
      const int k0 = s * 32 + half * 8;
      float4 f0 = *(const float4*)(zrow + k0);
      float4 f1 = *(const float4*)(zrow + k0 + 4);
      float4 g0 = *(const float4*)(zrow + k0 + 16);
      float4 g1 = *(const float4*)(zrow + k0 + 20);
      float xs[16] = {f0.x, f0.y, f0.z, f0.w, f1.x, f1.y, f1.z, f1.w,
                      g0.x, g0.y, g0.z, g0.w, g1.x, g1.y, g1.z, g1.w};
#pragma unroll
      for (int e = 0; e < 16; ++e) {
        __bf16 h = (__bf16)xs[e];
        a_hi[s][e] = h;
        a_lo[s][e] = (__bf16)(xs[e] - (float)h);
      }
    }
  }

  float minv[8];
  int   mini[8];
#pragma unroll
  for (int r = 0; r < 8; ++r) { minv[r] = 3.402823e38f; mini[r] = 0; }

  wait_asynccnt0();
  __syncthreads();

  for (int ch = 0; ch < N_CHUNKS; ++ch) {
    const int cur = ch & 1;
    // ---- Kick off async DMA for the next chunk into the other buffer.
    if (ch + 1 < N_CHUNKS) {
      const size_t grow = (size_t)((ch + 1) * N_CHUNK + srow) * D_DIM + scol;
      const __bf16* ghi = cb_hi + grow;
      const __bf16* glo = cb_lo + grow;
      const unsigned int dhi = cur ? lds_hi0 : lds_hi1;
      const unsigned int dlo = cur ? lds_lo0 : lds_lo1;
#pragma unroll
      for (int j = 0; j < 64; j += 8) {
        async_load_b128(dhi + j * 2, ghi + j);
        async_load_b128(dlo + j * 2, glo + j);
      }
    }

    // ---- 4 WMMA tiles of 16 codes; every wave scans all 64 codes.
    for (int t = 0; t < N_CHUNK / 16; ++t) {
      const int nloc = t * 16 + mlane;                // B column = code row
      const __bf16* bh = &s_hi[cur][nloc * CB_STRIDE];
      const __bf16* bl = &s_lo[cur][nloc * CB_STRIDE];
      v8f acc = {};
#pragma unroll
      for (int s = 0; s < 8; ++s) {
        // ISA 16-bit B 32x16 layout: lane holds col n=lane%16;
        // element e -> k = half*8 + (e<8 ? e : e+8) (matches A).
        const int kb = s * 32 + half * 8;
        v8bf h0 = *(const v8bf*)(bh + kb);
        v8bf h1 = *(const v8bf*)(bh + kb + 16);
        v8bf l0 = *(const v8bf*)(bl + kb);
        v8bf l1 = *(const v8bf*)(bl + kb + 16);
        v16bf bhi = __builtin_shufflevector(h0, h1, 0, 1, 2, 3, 4, 5, 6, 7,
                                            8, 9, 10, 11, 12, 13, 14, 15);
        v16bf blo = __builtin_shufflevector(l0, l1, 0, 1, 2, 3, 4, 5, 6, 7,
                                            8, 9, 10, 11, 12, 13, 14, 15);
        acc = wmma_bf16(a_hi[s], bhi, acc);   // hi*hi
        acc = wmma_bf16(a_hi[s], blo, acc);   // hi*lo
        acc = wmma_bf16(a_lo[s], bhi, acc);   // lo*hi
      }
      const int nglob = ch * N_CHUNK + t * 16 + mlane;
      const float cs  = s_csq[nglob];
      // C/D layout: VGPR r, lane -> (m = r + 8*half, n = lane%16).
#pragma unroll
      for (int r = 0; r < 8; ++r) {
        float score = cs - 2.0f * acc[r];
        if (score < minv[r]) { minv[r] = score; mini[r] = nglob; }
      }
    }

    // Next chunk's DMA must land and all waves must be done reading
    // before buffers swap.
    wait_asynccnt0();
    __syncthreads();
  }

  // ---- Reduce (min, idx) across the 16 lanes of each half-group.
#pragma unroll
  for (int mask = 1; mask <= 8; mask <<= 1) {
#pragma unroll
    for (int r = 0; r < 8; ++r) {
      float ov = __shfl_xor(minv[r], mask, 32);
      int   oi = __shfl_xor(mini[r], mask, 32);
      if (ov < minv[r] || (ov == minv[r] && oi < mini[r])) {
        minv[r] = ov; mini[r] = oi;
      }
    }
  }
  if (mlane == 0) {
#pragma unroll
    for (int r = 0; r < 8; ++r) s_idx[wave * 16 + half * 8 + r] = mini[r];
  }
  __syncthreads();

  // ---- Gather: out = z + (cb[idx] - z)  (match reference STE arithmetic).
  {
    const int row = tid >> 1;           // 0..127 (token within block)
    const int c0  = (tid & 1) * 128;    // half a row each
    const int idx = s_idx[row];
    const float* src = cb_f32 + (size_t)idx * D_DIM + c0;
    const float* zr  = z + (size_t)(tok_base + row) * D_DIM + c0;
    float* dst = out + (size_t)(tok_base + row) * D_DIM + c0;
#pragma unroll 4
    for (int j = 0; j < 128; j += 4) {
      float4 c = *(const float4*)(src + j);
      float4 a = *(const float4*)(zr + j);
      float4 o;
      o.x = a.x + (c.x - a.x);
      o.y = a.y + (c.y - a.y);
      o.z = a.z + (c.z - a.z);
      o.w = a.w + (c.w - a.w);
      *(float4*)(dst + j) = o;
    }
  }
}

// ---------------------------------------------------------------------------
extern "C" void kernel_launch(void* const* d_in, const int* in_sizes, int n_in,
                              void* d_out, int out_size, void* d_ws, size_t ws_size,
                              hipStream_t stream) {
  (void)in_sizes; (void)n_in; (void)out_size; (void)ws_size;
  const float* z  = (const float*)d_in[0];   // [16,2048,256] fp32
  const float* cb = (const float*)d_in[1];   // [1024,256] fp32
  float* out = (float*)d_out;                // [16,2048,256] fp32

  char* ws = (char*)d_ws;
  __bf16* cb_hi = (__bf16*)(ws);                         // 512 KB
  __bf16* cb_lo = (__bf16*)(ws + (size_t)512 * 1024);    // 512 KB
  float*  csq   = (float*)(ws + (size_t)1024 * 1024);    // 4 KB

  vq_prep<<<K_CODES, 256, 0, stream>>>(cb, cb_hi, cb_lo, csq);
  vq_assign<<<(16 * 2048) / M_TILE, 256, 0, stream>>>(z, cb, cb_hi, cb_lo,
                                                      csq, out);
}
